// SpaceTemporalAttention_68788196213257
// MI455X (gfx1250) — matrix operand order repaired and checked by reading, not compiled, not run on hardware
//
#include <hip/hip_runtime.h>

typedef __attribute__((ext_vector_type(16))) _Float16 v16h;
typedef __attribute__((ext_vector_type(8)))  _Float16 v8h;
typedef __attribute__((ext_vector_type(2)))  _Float16 v2h;
typedef __attribute__((ext_vector_type(8)))  float    v8f;
typedef __attribute__((ext_vector_type(4)))  float    v4f;
typedef float f32x4u __attribute__((ext_vector_type(4), aligned(4)));

#define N_SEQ   2048
#define HEADS   16
#define DHEAD   64
#define DIM     1024
#define QKV_LD  3072

// Join two v8h into a v16h fragment register block
__device__ __forceinline__ v16h join8(v8h a, v8h b) {
    return __builtin_shufflevector(a, b, 0, 1, 2, 3, 4, 5, 6, 7,
                                         8, 9, 10, 11, 12, 13, 14, 15);
}

// ---------------------------------------------------------------------------
// Elementwise conversion / packing kernels
// ---------------------------------------------------------------------------
__global__ void cvt_f32_to_f16_kernel(const float* __restrict__ src,
                                      _Float16* __restrict__ dst, int n) {
    int i = blockIdx.x * blockDim.x + threadIdx.x;
    if (i < n) dst[i] = (_Float16)src[i];
}

__global__ void pack_wqkv_kernel(const float* __restrict__ Wq,
                                 const float* __restrict__ Wkv,
                                 _Float16* __restrict__ dst) {
    int i = blockIdx.x * blockDim.x + threadIdx.x;   // over 1024*3072
    if (i >= DIM * QKV_LD) return;
    int r = i / QKV_LD, c = i % QKV_LD;
    float v = (c < DIM) ? Wq[r * DIM + c] : Wkv[r * 2048 + (c - DIM)];
    dst[i] = (_Float16)v;
}

// T5 relative position bias table: tab[h * 4095 + (j - i + 2047)]
__global__ void bias_table_kernel(const float* __restrict__ rel_emb,
                                  float* __restrict__ tab) {
    int i = blockIdx.x * blockDim.x + threadIdx.x;   // 16 * 4095
    if (i >= HEADS * (2 * N_SEQ - 1)) return;
    int h = i / (2 * N_SEQ - 1);
    int t = i % (2 * N_SEQ - 1);
    int rel = t - (N_SEQ - 1);     // rel = j - i (k_pos - q_pos)
    int nn  = -rel;                // n = -relative_position
    int ret = 0;
    if (nn < 0) { ret = 16; nn = -nn; }
    int bucket;
    if (nn < 8) {
        bucket = ret + nn;
    } else {
        float lv = logf((float)nn * 0.125f) / 2.7725887f * 8.0f;
        int val = 8 + (int)lv;
        if (val > 15) val = 15;
        bucket = ret + val;
    }
    tab[h * (2 * N_SEQ - 1) + t] = rel_emb[bucket * HEADS + h];
}

// ---------------------------------------------------------------------------
// CDNA5 wave32 WMMA 16x16x32 f16 fragment layouts:
//   A frag : lane holds row (lane&15); elems = K {hi*8..+7} and {16+hi*8..+7}
//            -> two contiguous v8h from a row-major tile
//   B frag : lane holds col (lane&15); elems = K {hi*8..+7} and {16+hi*8..+7}
//            -> two contiguous v8h from a SWIZZLED [colTile][col][K] tile
//   C/D    : lane L, VGPR v -> row = v + (L>>4)*8, col = (L&15)
// ---------------------------------------------------------------------------
#define LDA_S  40        // A LDS row stride (halves)
#define SWB_S  40        // swizzled-B per-column stride (halves)
#define SWB_T  640       // swizzled-B per-16-col tile (16 * 40)

// ---------------------------------------------------------------------------
// Generic f16 GEMM (fp32 accumulate): C[M x N] = A[M x K] * B[K x N] (+ bias)
// Block = 256 threads (8 waves). C tile = 128 x 64, each wave owns 32 x 32.
// Software-pipelined: next K-tile global loads overlap WMMA compute.
// ---------------------------------------------------------------------------
template<bool OUT_F32, bool ADD_BIAS>
__global__ __launch_bounds__(256)
void gemm_f16_kernel(const _Float16* __restrict__ A, const _Float16* __restrict__ B,
                     void* __restrict__ Cout, const float* __restrict__ bias,
                     int K, int lda, int ldb, int ldc) {
    __shared__ alignas(16) _Float16 As[128 * LDA_S];
    __shared__ alignas(16) _Float16 Bsw[4 * SWB_T];

    const int tid  = threadIdx.x;
    const int lane = tid & 31;
    const int w    = tid >> 5;
    const int hi   = lane >> 4;
    const int lo   = lane & 15;
    const int m0   = blockIdx.y * 128;
    const int n0   = blockIdx.x * 64;
    const int mBase = (w >> 1) * 32;
    const int nBase = (w & 1) * 32;

    // A staging: each thread owns rows rA, rA+64 at col group cA (8 halves)
    const int rA = tid >> 2;
    const int cA = (tid & 3) * 8;
    // B staging: threads 0..127 own K-rows rB, rB+1 at col group cB (8 halves)
    const int rB = (tid >> 3) * 2;
    const int cB = (tid & 7) * 8;
    const int bSwBase = (cB >> 4) * SWB_T + (cB & 15) * SWB_S + rB;

    v8f acc[2][2] = {};

    // Preload first K-tile into registers
    v8h a0 = *(const v8h*)&A[(size_t)(m0 + rA) * lda + cA];
    v8h a1 = *(const v8h*)&A[(size_t)(m0 + rA + 64) * lda + cA];
    v8h b0, b1;
    if (tid < 128) {
        b0 = *(const v8h*)&B[(size_t)rB * ldb + n0 + cB];
        b1 = *(const v8h*)&B[(size_t)(rB + 1) * ldb + n0 + cB];
    }

    for (int k0 = 0; k0 < K; k0 += 32) {
        // Commit staged registers to LDS (B goes in swizzled layout)
        *(v8h*)&As[rA * LDA_S + cA]        = a0;
        *(v8h*)&As[(rA + 64) * LDA_S + cA] = a1;
        if (tid < 128) {
#pragma unroll
            for (int j = 0; j < 8; ++j) {
                v2h p; p[0] = b0[j]; p[1] = b1[j];
                *(v2h*)&Bsw[bSwBase + j * SWB_S] = p;
            }
        }
        __syncthreads();

        // Issue next tile's global loads (results not needed until next iter)
        if (k0 + 32 < K) {
            a0 = *(const v8h*)&A[(size_t)(m0 + rA) * lda + (k0 + 32) + cA];
            a1 = *(const v8h*)&A[(size_t)(m0 + rA + 64) * lda + (k0 + 32) + cA];
            if (tid < 128) {
                b0 = *(const v8h*)&B[(size_t)(k0 + 32 + rB) * ldb + n0 + cB];
                b1 = *(const v8h*)&B[(size_t)(k0 + 32 + rB + 1) * ldb + n0 + cB];
            }
        }

        // Fragment loads: all b128 pairs
        v16h af[2], bf[2];
#pragma unroll
        for (int i = 0; i < 2; ++i) {
            const _Float16* p = &As[(mBase + i * 16 + lo) * LDA_S];
            af[i] = join8(*(const v8h*)(p + hi * 8),
                          *(const v8h*)(p + 16 + hi * 8));
        }
#pragma unroll
        for (int j = 0; j < 2; ++j) {
            const _Float16* p = &Bsw[((nBase >> 4) + j) * SWB_T + lo * SWB_S + hi * 8];
            bf[j] = join8(*(const v8h*)p, *(const v8h*)(p + 16));
        }

#pragma unroll
        for (int i = 0; i < 2; ++i)
#pragma unroll
            for (int j = 0; j < 2; ++j)
                acc[i][j] = __builtin_amdgcn_wmma_f32_16x16x32_f16(
                    false, af[i], false, bf[j], (short)0, acc[i][j], false, false);
        __syncthreads();
    }

    // Write out C tile
#pragma unroll
    for (int i = 0; i < 2; ++i) {
#pragma unroll
        for (int j = 0; j < 2; ++j) {
            int col = n0 + nBase + j * 16 + lo;
#pragma unroll
            for (int v = 0; v < 8; ++v) {
                int row = m0 + mBase + i * 16 + v + hi * 8;
                float val = acc[i][j][v];
                if (ADD_BIAS) val += bias[col];
                if (OUT_F32)
                    ((float*)Cout)[(size_t)row * ldc + col] = val;
                else
                    ((_Float16*)Cout)[(size_t)row * ldc + col] = (_Float16)val;
            }
        }
    }
}

// ---------------------------------------------------------------------------
// Flash attention: one block per (64-query tile, b*h). 128 threads = 4 waves,
// each wave owns 16 query rows. K staged row-major (K^T frag is contiguous),
// V staged swizzled (PV B-frags are b128 loads). Softmax done in row-per-lane
// layout via a per-wave f32 LDS round trip: per-row stats are per-lane
// scalars, reductions need a single shfl_xor(16), and the exp'd scores come
// out directly in A-fragment element order (no separate f16 P scratch).
// ---------------------------------------------------------------------------
#define KV_S 72
#define SS_S 36      // f32 score-scratch row stride

__global__ __launch_bounds__(128)
void attn_kernel(const _Float16* __restrict__ qkv, const float* __restrict__ btab,
                 _Float16* __restrict__ Oh) {
    __shared__ alignas(16) _Float16 Ks[32 * KV_S];
    __shared__ alignas(16) _Float16 Vsw[4 * SWB_T];
    __shared__ alignas(16) float    Ss[4][16 * SS_S];

    const int tid  = threadIdx.x;
    const int lane = tid & 31;
    const int w    = tid >> 5;
    const int hi   = lane >> 4;
    const int lo   = lane & 15;
    const int bh = blockIdx.y;
    const int b  = bh >> 4;
    const int h  = bh & 15;
    const int q0 = blockIdx.x * 64 + w * 16;   // first query row of this wave
    const float scale = 0.125f;                // 64^-0.5

    // Staging: thread owns key rows rK, rK+1 at d-col group cK (8 halves)
    const int rK = (tid >> 3) * 2;
    const int cK = (tid & 7) * 8;
    const int vSwBase = (cK >> 4) * SWB_T + (cK & 15) * SWB_S + rK;

    // Preload Q fragments for this wave's 16 rows (2 K-chunks of d)
    v16h qf[2];
    {
        const _Float16* qp = &qkv[(size_t)(b * N_SEQ + q0 + lo) * QKV_LD + h * DHEAD];
#pragma unroll
        for (int dc = 0; dc < 2; ++dc)
            qf[dc] = join8(*(const v8h*)(qp + dc * 32 + hi * 8),
                           *(const v8h*)(qp + dc * 32 + 16 + hi * 8));
    }

    // Per-lane softmax state for query row (q0 + lo), replicated in both halves
    float mrow = -1e30f, lrow = 0.0f;
    v8f acc[4] = {};

    const float* brow = &btab[h * (2 * N_SEQ - 1)];
    float* sw = &Ss[w][0];

    // Preload first K/V block
    v8h kR0, kR1, vR0, vR1;
    {
        size_t g = (size_t)(b * N_SEQ + rK) * QKV_LD + h * DHEAD + cK;
        kR0 = *(const v8h*)&qkv[g + 1024];
        kR1 = *(const v8h*)&qkv[g + 1024 + QKV_LD];
        vR0 = *(const v8h*)&qkv[g + 2048];
        vR1 = *(const v8h*)&qkv[g + 2048 + QKV_LD];
    }

    for (int j0 = 0; j0 < N_SEQ; j0 += 32) {
        // Commit staged K (row-major) and V (swizzled) blocks to LDS
        *(v8h*)&Ks[rK * KV_S + cK]       = kR0;
        *(v8h*)&Ks[(rK + 1) * KV_S + cK] = kR1;
#pragma unroll
        for (int j = 0; j < 8; ++j) {
            v2h p; p[0] = vR0[j]; p[1] = vR1[j];
            *(v2h*)&Vsw[vSwBase + j * SWB_S] = p;
        }
        __syncthreads();

        // Issue next block's global loads
        if (j0 + 32 < N_SEQ) {
            size_t g = (size_t)(b * N_SEQ + j0 + 32 + rK) * QKV_LD + h * DHEAD + cK;
            kR0 = *(const v8h*)&qkv[g + 1024];
            kR1 = *(const v8h*)&qkv[g + 1024 + QKV_LD];
            vR0 = *(const v8h*)&qkv[g + 2048];
            vR1 = *(const v8h*)&qkv[g + 2048 + QKV_LD];
        }

        // S = Q K^T for 16 x 32 scores (2 column fragments, C layout)
#pragma unroll
        for (int jf = 0; jf < 2; ++jf) {
            v8f s = {};
#pragma unroll
            for (int dc = 0; dc < 2; ++dc) {
                const _Float16* p = &Ks[(jf * 16 + lo) * KV_S + dc * 32 + hi * 8];
                v16h bf = join8(*(const v8h*)p, *(const v8h*)(p + 16));
                s = __builtin_amdgcn_wmma_f32_16x16x32_f16(
                    false, qf[dc], false, bf, (short)0, s, false, false);
            }
            // Spill raw scores to per-wave scratch (C layout addresses)
#pragma unroll
            for (int v = 0; v < 8; ++v)
                sw[(v + hi * 8) * SS_S + jf * 16 + lo] = s[v];
        }

        // Re-read in row-per-lane layout: lane holds row lo, 16 cols in exact
        // A-fragment element order (same-wave LDS ops are in-order).
        const float* sp = &sw[lo * SS_S];
        v4f r0 = *(const v4f*)(sp + hi * 8);
        v4f r1 = *(const v4f*)(sp + hi * 8 + 4);
        v4f r2 = *(const v4f*)(sp + 16 + hi * 8);
        v4f r3 = *(const v4f*)(sp + 16 + hi * 8 + 4);

        // Bias: contiguous runs, idx = (j0 + kk) - (q0 + lo) + 2047
        const float* bp = brow + (j0 - (q0 + lo) + (N_SEQ - 1));
        f32x4u bb0 = *(const f32x4u*)(bp + hi * 8);
        f32x4u bb1 = *(const f32x4u*)(bp + hi * 8 + 4);
        f32x4u bb2 = *(const f32x4u*)(bp + 16 + hi * 8);
        f32x4u bb3 = *(const f32x4u*)(bp + 16 + hi * 8 + 4);

        float p[16];
#pragma unroll
        for (int e = 0; e < 4; ++e) {
            p[e]      = fmaf(r0[e], scale, bb0[e]);
            p[4 + e]  = fmaf(r1[e], scale, bb1[e]);
            p[8 + e]  = fmaf(r2[e], scale, bb2[e]);
            p[12 + e] = fmaf(r3[e], scale, bb3[e]);
        }

        // Row max: per-lane tree + one cross-half shuffle
        float mx = p[0];
#pragma unroll
        for (int e = 1; e < 16; ++e) mx = fmaxf(mx, p[e]);
        mx = fmaxf(mx, __shfl_xor(mx, 16, 32));
        float mn   = fmaxf(mrow, mx);
        float corr = __expf(mrow - mn);

        float rs = 0.0f;
#pragma unroll
        for (int e = 0; e < 16; ++e) {
            p[e] = __expf(p[e] - mn);
            rs += p[e];
        }
        rs += __shfl_xor(rs, 16, 32);
        lrow = lrow * corr + rs;
        mrow = mn;

        // P A-fragment directly from registers
        v16h pf;
#pragma unroll
        for (int e = 0; e < 16; ++e) pf[e] = (_Float16)p[e];

        // Broadcast per-row rescale factors to C-layout rows (v + hi*8)
        float corr_c[8];
#pragma unroll
        for (int v = 0; v < 8; ++v) corr_c[v] = __shfl(corr, v + hi * 8, 32);
#pragma unroll
        for (int c = 0; c < 4; ++c)
#pragma unroll
            for (int v = 0; v < 8; ++v)
                acc[c][v] *= corr_c[v];

        // O += P V (4 output-column fragments, V frags are b128 pairs)
#pragma unroll
        for (int c = 0; c < 4; ++c) {
            const _Float16* pv = &Vsw[c * SWB_T + lo * SWB_S + hi * 8];
            v16h bf = join8(*(const v8h*)pv, *(const v8h*)(pv + 16));
            acc[c] = __builtin_amdgcn_wmma_f32_16x16x32_f16(
                false, pf, false, bf, (short)0, acc[c], false, false);
        }
        __syncthreads();
    }

    // Normalize (broadcast l to C-layout rows) and write O (b*n, 1024) f16
    float linv[8];
#pragma unroll
    for (int v = 0; v < 8; ++v) linv[v] = 1.0f / __shfl(lrow, v + hi * 8, 32);
#pragma unroll
    for (int c = 0; c < 4; ++c) {
#pragma unroll
        for (int v = 0; v < 8; ++v) {
            float val = acc[c][v] * linv[v];
            Oh[(size_t)(b * N_SEQ + q0 + v + hi * 8) * DIM +
               h * DHEAD + c * 16 + lo] = (_Float16)val;
        }
    }
}

// ---------------------------------------------------------------------------
// Launch: cast -> pack -> bias table -> QKV GEMM -> attention -> out GEMM
// ---------------------------------------------------------------------------
extern "C" void kernel_launch(void* const* d_in, const int* in_sizes, int n_in,
                              void* d_out, int out_size, void* d_ws, size_t ws_size,
                              hipStream_t stream) {
    const float* x       = (const float*)d_in[0];
    const float* Wq      = (const float*)d_in[1];
    const float* Wkv     = (const float*)d_in[2];
    const float* Wo      = (const float*)d_in[3];
    const float* bo      = (const float*)d_in[4];
    const float* rel_emb = (const float*)d_in[5];

    char* ws = (char*)d_ws;
    _Float16* xh   = (_Float16*)(ws + 0);          //  8 MB : x as f16 (4096 x 1024)
    _Float16* wqkv = (_Float16*)(ws + 8388608);    //  6 MB : [Wq | Wkv] f16 (1024 x 3072)
    _Float16* woh  = (_Float16*)(ws + 14680064);   //  2 MB : Wo f16 (1024 x 1024)
    _Float16* qkvb = (_Float16*)(ws + 16777216);   // 24 MB : QKV f16 (4096 x 3072)
    _Float16* oh   = (_Float16*)(ws + 41943040);   //  8 MB : attn out f16 (4096 x 1024)
    float*    btab = (float*)(ws + 50331648);      // 256KB : bias table (16 x 4095)

    const int nX  = 2 * N_SEQ * DIM;     // 4194304
    const int nWp = DIM * QKV_LD;        // 3145728
    const int nWo = DIM * DIM;           // 1048576
    const int nB  = HEADS * (2 * N_SEQ - 1);

    cvt_f32_to_f16_kernel<<<(nX + 255) / 256, 256, 0, stream>>>(x, xh, nX);
    pack_wqkv_kernel<<<(nWp + 255) / 256, 256, 0, stream>>>(Wq, Wkv, wqkv);
    cvt_f32_to_f16_kernel<<<(nWo + 255) / 256, 256, 0, stream>>>(Wo, woh, nWo);
    bias_table_kernel<<<(nB + 255) / 256, 256, 0, stream>>>(rel_emb, btab);

    // QKV projection: (4096 x 1024) * (1024 x 3072) -> f16 QKV
    gemm_f16_kernel<false, false><<<dim3(QKV_LD / 64, (2 * N_SEQ) / 128), 256, 0, stream>>>(
        xh, wqkv, (void*)qkvb, nullptr, DIM, DIM, QKV_LD, QKV_LD);

    // Attention: grid (query tiles, b*h)
    attn_kernel<<<dim3(N_SEQ / 64, 2 * HEADS), 128, 0, stream>>>(qkvb, btab, oh);

    // Output projection + bias: (4096 x 1024) * (1024 x 1024) + bo -> fp32 out
    gemm_f16_kernel<true, true><<<dim3(DIM / 64, (2 * N_SEQ) / 128), 256, 0, stream>>>(
        oh, woh, d_out, bo, DIM, DIM, DIM, DIM);
}
